// BigBirdAttention_37915971289180
// MI455X (gfx1250) — matrix operand order repaired and checked by reading, not compiled, not run on hardware
//
#include <hip/hip_runtime.h>

// CDNA5 / gfx1250: wave32, WMMA 16x16x32 bf16 -> f32
typedef __bf16 v16bf __attribute__((ext_vector_type(16)));
typedef float  v8f   __attribute__((ext_vector_type(8)));

#define DEV __device__ __forceinline__

DEV int lane_id() { return (int)(threadIdx.x & 31); }

DEV v8f wmma_bf16(v16bf a, v16bf b, v8f c) {
  // (neg_a, A, neg_b, B, c_mod, C, reuse_a, reuse_b)
  return __builtin_amdgcn_wmma_f32_16x16x32_bf16(false, a, false, b, (short)0, c, false, false);
}

// ---- CDNA5 async copy: global -> LDS, 16 bytes per lane, ASYNCcnt-tracked ----
DEV void async_copy_b128(float* lds_dst, const float* gsrc) {
  unsigned ldsoff = (unsigned)(size_t)lds_dst;   // low 32 bits of LDS aperture addr = LDS offset
  asm volatile("global_load_async_to_lds_b128 %0, %1, off"
               :: "v"(ldsoff), "v"(gsrc) : "memory");
}
DEV void wait_async0() { asm volatile("s_wait_asynccnt 0x0" ::: "memory"); }

// ---- Tile loaders (this code's k order k = hi*16 + e in A and B alike, not the ISA table order; wave32) ----
// A 16x32 (MxK): lane (m,hi): element e -> k = hi*16 + e
DEV v16bf load_a_f32(const float* base, int ld) {   // base -> (row0,k0), row-major
  int l = lane_id(); int m = l & 15; int hi = l >> 4;
  const float* r = base + (size_t)m * ld + hi * 16;
  v16bf a;
#pragma unroll
  for (int e = 0; e < 16; ++e) a[e] = (__bf16)r[e];
  return a;
}

DEV v16bf load_a_bf(const __bf16* base, int ld) {   // same, bf16 source (LDS)
  int l = lane_id(); int m = l & 15; int hi = l >> 4;
  const __bf16* r = base + (size_t)m * ld + hi * 16;
  v16bf a;
#pragma unroll
  for (int e = 0; e < 16; ++e) a[e] = r[e];
  return a;
}

// B 32x16 (KxN): lane (n,hi): element e -> k = hi*16 + e
// Transposed source: B[k][n] = srcT[n][k]; base -> (n0,k0) of srcT => contiguous run of 16
DEV v16bf load_bT_f32(const float* base, int ld) {
  int l = lane_id(); int n = l & 15; int hi = l >> 4;
  const float* r = base + (size_t)n * ld + hi * 16;
  v16bf b;
#pragma unroll
  for (int e = 0; e < 16; ++e) b[e] = (__bf16)r[e];
  return b;
}

// Transposed source with indirection: column n is row kidx[j0+n] (64-wide K/V rows)
DEV v16bf load_bT_idx(const float* src, const int* kidx, int j0, int d0) {
  int l = lane_id(); int n = l & 15; int hi = l >> 4;
  const float* r = src + (size_t)kidx[j0 + n] * 64 + d0 + hi * 16;
  v16bf b;
#pragma unroll
  for (int e = 0; e < 16; ++e) b[e] = (__bf16)r[e];
  return b;
}

// Transposed source, global-column keys: j<64 -> j (block 0), else N-128+j (block nb-1)
DEV v16bf load_bT_gc(const float* src, int j0, int d0) {
  int l = lane_id(); int n = l & 15; int hi = l >> 4;
  int j = j0 + n;
  int kg = (j < 64) ? j : (4096 - 128 + j);
  const float* r = src + (size_t)kg * 64 + d0 + hi * 16;
  v16bf b;
#pragma unroll
  for (int e = 0; e < 16; ++e) b[e] = (__bf16)r[e];
  return b;
}

// Row-major B from LDS (rows = K, stride ld): strided gather, fp32 source
DEV v16bf load_b_lds(const float* src, int ld, int k0, int n0) {
  int l = lane_id(); int n = l & 15; int hi = l >> 4;
  v16bf b;
#pragma unroll
  for (int e = 0; e < 16; ++e)
    b[e] = (__bf16)src[(k0 + hi * 16 + e) * ld + n0 + n];
  return b;
}

DEV v16bf load_b_idx(const float* src, const int* kidx, int k0, int n0) {
  int l = lane_id(); int n = l & 15; int hi = l >> 4;
  v16bf b;
#pragma unroll
  for (int e = 0; e < 16; ++e)
    b[e] = (__bf16)src[(size_t)kidx[k0 + hi * 16 + e] * 64 + n0 + n];
  return b;
}

DEV v16bf load_b_gc(const float* src, int k0, int n0) {
  int l = lane_id(); int n = l & 15; int hi = l >> 4;
  v16bf b;
#pragma unroll
  for (int e = 0; e < 16; ++e) {
    int j = k0 + hi * 16 + e;
    int kg = (j < 64) ? j : (4096 - 128 + j);
    b[e] = (__bf16)src[(size_t)kg * 64 + n0 + n];
  }
  return b;
}

// C/D 16x16 f32: lane (n,hi), VGPR v holds row m = hi*8 + v, col n

// =====================================================================
// Kernel 1: Y = X @ W^T + bias.  X: (16384,1024), W: (1024,1024).
// permute=1: write (B,H,N,dh) layout; permute=0: plain (token, 1024).
// Block = 256 threads (8 waves); each wave does a 16x64 strip.
// grid = (16384/128, 1024/64)
// =====================================================================
__global__ void bigbird_proj_kernel(const float* __restrict__ src,
                                    const float* __restrict__ W,
                                    const float* __restrict__ bias,
                                    float* __restrict__ dst, int permute) {
  int wave = threadIdx.x >> 5;
  int row0 = blockIdx.x * 128 + wave * 16;
  int col0 = blockIdx.y * 64;

  v8f acc[4] = {v8f{}, v8f{}, v8f{}, v8f{}};
  for (int k0 = 0; k0 < 1024; k0 += 32) {
    if (k0 + 128 < 1024)   // stream-ahead prefetch of the A operand
      __builtin_prefetch(src + (size_t)row0 * 1024 + k0 + 128, 0, 1);
    v16bf a = load_a_f32(src + (size_t)row0 * 1024 + k0, 1024);
#pragma unroll
    for (int t = 0; t < 4; ++t) {
      v16bf bm = load_bT_f32(W + (size_t)(col0 + 16 * t) * 1024 + k0, 1024);
      acc[t] = wmma_bf16(a, bm, acc[t]);
    }
  }

  int l = lane_id(); int n = l & 15; int hi = l >> 4;
#pragma unroll
  for (int t = 0; t < 4; ++t) {
    int col = col0 + 16 * t + n;
    float bv = bias[col];
#pragma unroll
    for (int v = 0; v < 8; ++v) {
      int token = row0 + hi * 8 + v;
      float val = acc[t][v] + bv;
      if (permute) {
        int b_i = token >> 12;        // N = 4096
        int n_i = token & 4095;
        int h = col >> 6;             // dh = 64
        int d = col & 63;
        dst[(((size_t)(b_i * 16 + h) * 4096) + n_i) * 64 + d] = val;
      } else {
        dst[(size_t)token * 1024 + col] = val;
      }
    }
  }
}

// =====================================================================
// Kernel 2: local window (192 keys) + global-column (128 keys) attention.
// One block per (query block, b*h). 256 threads. 32 queries per half.
// =====================================================================
__global__ void bigbird_local_kernel(const float* __restrict__ q,
                                     const float* __restrict__ k,
                                     const float* __restrict__ v,
                                     float* __restrict__ ctx) {
  int nblk = blockIdx.x;                 // 0..63
  int bh   = blockIdx.y;                 // 0..63
  const float* qbh = q + (size_t)bh * 4096 * 64;
  const float* kbh = k + (size_t)bh * 4096 * 64;
  const float* vbh = v + (size_t)bh * 4096 * 64;

  __shared__ float  s_s[32 * 192];
  __shared__ __bf16 s_p[32 * 192];
  __shared__ int    s_kidx[192];

  int tid = threadIdx.x, wave = tid >> 5, l = tid & 31;
  int n = l & 15, hi = l >> 4;

  if (tid < 192) {
    int w = tid >> 6;                                   // window part 0,1,2
    int blk = (nblk + w - 1 + 64) & 63;                 // jnp.roll wrap
    s_kidx[tid] = blk * 64 + (tid & 63);
  }
  __syncthreads();

  int b_i = bh >> 4, h = bh & 15;
  int mt = wave & 1, nt4 = wave >> 1;                   // output tile of this wave

  for (int half = 0; half < 2; ++half) {
    int q0 = nblk * 64 + half * 32;                     // absolute first query row

    // ---- phase 1 scores: 32x192, 24 tiles (2x12), 3 per wave ----
#pragma unroll
    for (int i = 0; i < 3; ++i) {
      int t = wave + 8 * i;
      int smt = t & 1, snt = t >> 1;
      v8f acc = {};
      for (int d0 = 0; d0 < 64; d0 += 32) {
        v16bf a  = load_a_f32(qbh + (size_t)(q0 + smt * 16) * 64 + d0, 64);
        v16bf bm = load_bT_idx(kbh, s_kidx, snt * 16, d0);
        acc = wmma_bf16(a, bm, acc);
      }
#pragma unroll
      for (int vv = 0; vv < 8; ++vv)
        s_s[(smt * 16 + hi * 8 + vv) * 192 + snt * 16 + n] = acc[vv] * 0.125f;
    }
    __syncthreads();

    // ---- softmax over 192 (fp32), one thread per row ----
    if (tid < 32) {
      float m = -1e30f;
      for (int j = 0; j < 192; ++j) m = fmaxf(m, s_s[tid * 192 + j]);
      float ssum = 0.f;
      for (int j = 0; j < 192; ++j) {
        float e = __expf(s_s[tid * 192 + j] - m);
        s_s[tid * 192 + j] = e;
        ssum += e;
      }
      float inv = 1.f / ssum;
      for (int j = 0; j < 192; ++j) s_p[tid * 192 + j] = (__bf16)(s_s[tid * 192 + j] * inv);
    }
    __syncthreads();

    // ---- O1 = P @ V_window : 32x64, 8 tiles, one per wave ----
    v8f oacc = {};
    for (int k0 = 0; k0 < 192; k0 += 32) {
      v16bf a  = load_a_bf(s_p + (size_t)(mt * 16) * 192 + k0, 192);
      v16bf bm = load_b_idx(vbh, s_kidx, k0, nt4 * 16);
      oacc = wmma_bf16(a, bm, oacc);
    }
    __syncthreads();

    // ---- phase 2 scores: 32x128 vs global keys, 16 tiles, 2 per wave ----
#pragma unroll
    for (int i = 0; i < 2; ++i) {
      int t = wave + 8 * i;
      int smt = t & 1, snt = t >> 1;
      v8f acc = {};
      for (int d0 = 0; d0 < 64; d0 += 32) {
        v16bf a  = load_a_f32(qbh + (size_t)(q0 + smt * 16) * 64 + d0, 64);
        v16bf bm = load_bT_gc(kbh, snt * 16, d0);
        acc = wmma_bf16(a, bm, acc);
      }
#pragma unroll
      for (int vv = 0; vv < 8; ++vv)
        s_s[(smt * 16 + hi * 8 + vv) * 128 + snt * 16 + n] = acc[vv] * 0.125f;
    }
    __syncthreads();

    if (tid < 32) {
      float m = -1e30f;
      for (int j = 0; j < 128; ++j) m = fmaxf(m, s_s[tid * 128 + j]);
      float ssum = 0.f;
      for (int j = 0; j < 128; ++j) {
        float e = __expf(s_s[tid * 128 + j] - m);
        s_s[tid * 128 + j] = e;
        ssum += e;
      }
      float inv = 1.f / ssum;
      for (int j = 0; j < 128; ++j) s_p[tid * 128 + j] = (__bf16)(s_s[tid * 128 + j] * inv);
    }
    __syncthreads();

    // ---- O2 accumulated into oacc (out_local + out_gk) ----
    for (int k0 = 0; k0 < 128; k0 += 32) {
      v16bf a  = load_a_bf(s_p + (size_t)(mt * 16) * 128 + k0, 128);
      v16bf bm = load_b_gc(vbh, k0, nt4 * 16);
      oacc = wmma_bf16(a, bm, oacc);
    }

    // write ctx (B,N,D) layout: col = h*64 + d
#pragma unroll
    for (int vv = 0; vv < 8; ++vv) {
      int tok = q0 + mt * 16 + hi * 8 + vv;
      ctx[((size_t)b_i * 4096 + tok) * 1024 + h * 64 + nt4 * 16 + n] = oacc[vv];
    }
    __syncthreads();   // before next half reuses LDS
  }
}

// =====================================================================
// Kernel 3: global-row blocks: full attention over 4096 keys.
// Two-pass flash, 64-key chunks. One block per (gi, b*h). 256 threads.
// Pass 2 stages each V chunk via GLOBAL_LOAD_ASYNC_TO_LDS_B128 (ASYNCcnt),
// overlapped with the QK^T WMMA work, then gathers B-tiles from LDS.
// Overwrites ctx at the global query blocks (launched after kernel 2).
// =====================================================================
__global__ void bigbird_global_kernel(const float* __restrict__ q,
                                      const float* __restrict__ k,
                                      const float* __restrict__ v,
                                      const int* __restrict__ gidx,
                                      float* __restrict__ ctx) {
  int bh = blockIdx.y;
  int qb = gidx[blockIdx.x];
  int q0 = qb * 64;
  const float* qbh = q + (size_t)bh * 4096 * 64;
  const float* kbh = k + (size_t)bh * 4096 * 64;
  const float* vbh = v + (size_t)bh * 4096 * 64;

  __shared__ float  s_s[64 * 64];   // 16 KB scores
  __shared__ __bf16 s_p[64 * 64];   //  8 KB probabilities
  __shared__ float  s_v[64 * 64];   // 16 KB async-staged V chunk
  __shared__ float  s_l[64];

  int tid = threadIdx.x, wave = tid >> 5, l = tid & 31;
  int n = l & 15, hi = l >> 4;

  float mrun = -1e30f;     // rows owned by tid<64
  // ---- pass 1: row max ----
  for (int c = 0; c < 4096; c += 64) {
#pragma unroll
    for (int i = 0; i < 2; ++i) {
      int t = wave + 8 * i;                // 0..15 -> 4x4 tiles
      int smt = t & 3, snt = t >> 2;
      v8f acc = {};
      for (int d0 = 0; d0 < 64; d0 += 32) {
        v16bf a  = load_a_f32(qbh + (size_t)(q0 + smt * 16) * 64 + d0, 64);
        v16bf bm = load_bT_f32(kbh + (size_t)(c + snt * 16) * 64 + d0, 64);
        acc = wmma_bf16(a, bm, acc);
      }
#pragma unroll
      for (int vv = 0; vv < 8; ++vv)
        s_s[(smt * 16 + hi * 8 + vv) * 64 + snt * 16 + n] = acc[vv] * 0.125f;
    }
    __syncthreads();
    if (tid < 64)
      for (int j = 0; j < 64; ++j) mrun = fmaxf(mrun, s_s[tid * 64 + j]);
    __syncthreads();
  }

  // ---- pass 2: exp/sum + O accumulation ----
  float lrun = 0.f;
  v8f oacc[2] = {v8f{}, v8f{}};
  for (int c = 0; c < 4096; c += 64) {
    // kick off async DMA of this chunk's V (64x64 fp32 = 16 KB) into LDS,
    // overlapped with the QK^T WMMAs below
#pragma unroll
    for (int i = 0; i < 4; ++i) {
      int off = (i * 256 + tid) * 4;       // float index, 16B per lane
      async_copy_b128(&s_v[off], vbh + (size_t)c * 64 + off);
    }
#pragma unroll
    for (int i = 0; i < 2; ++i) {
      int t = wave + 8 * i;
      int smt = t & 3, snt = t >> 2;
      v8f acc = {};
      for (int d0 = 0; d0 < 64; d0 += 32) {
        v16bf a  = load_a_f32(qbh + (size_t)(q0 + smt * 16) * 64 + d0, 64);
        v16bf bm = load_bT_f32(kbh + (size_t)(c + snt * 16) * 64 + d0, 64);
        acc = wmma_bf16(a, bm, acc);
      }
#pragma unroll
      for (int vv = 0; vv < 8; ++vv)
        s_s[(smt * 16 + hi * 8 + vv) * 64 + snt * 16 + n] = acc[vv] * 0.125f;
    }
    wait_async0();           // this wave's V DMA complete (LDS visible)
    __syncthreads();         // all waves' scores + DMAs done
    if (tid < 64) {
      for (int j = 0; j < 64; ++j) {
        float e = __expf(s_s[tid * 64 + j] - mrun);
        lrun += e;
        s_p[tid * 64 + j] = (__bf16)e;
      }
    }
    __syncthreads();
#pragma unroll
    for (int i = 0; i < 2; ++i) {
      int t = wave + 8 * i;                // 0..15 -> 4x4 O tiles
      int omt = t & 3, ont = t >> 2;
      for (int k0 = 0; k0 < 64; k0 += 32) {
        v16bf a  = load_a_bf(s_p + (size_t)(omt * 16) * 64 + k0, 64);
        v16bf bm = load_b_lds(s_v, 64, k0, ont * 16);
        oacc[i] = wmma_bf16(a, bm, oacc[i]);
      }
    }
    __syncthreads();         // protect s_s/s_p/s_v before next chunk
  }

  if (tid < 64) s_l[tid] = 1.f / lrun;
  __syncthreads();

  int b_i = bh >> 4, h = bh & 15;
#pragma unroll
  for (int i = 0; i < 2; ++i) {
    int t = wave + 8 * i;
    int omt = t & 3, ont = t >> 2;
#pragma unroll
    for (int vv = 0; vv < 8; ++vv) {
      int row = omt * 16 + hi * 8 + vv;
      int tok = q0 + row;
      ctx[((size_t)b_i * 4096 + tok) * 1024 + h * 64 + ont * 16 + n] = oacc[i][vv] * s_l[row];
    }
  }
}

// =====================================================================
extern "C" void kernel_launch(void* const* d_in, const int* in_sizes, int n_in,
                              void* d_out, int out_size, void* d_ws, size_t ws_size,
                              hipStream_t stream) {
  (void)in_sizes; (void)n_in; (void)out_size; (void)ws_size;
  const float* x  = (const float*)d_in[0];
  const float* Wq = (const float*)d_in[1];
  const float* bq = (const float*)d_in[2];
  const float* Wk = (const float*)d_in[3];
  const float* bk = (const float*)d_in[4];
  const float* Wv = (const float*)d_in[5];
  const float* bv = (const float*)d_in[6];
  const float* Wo = (const float*)d_in[7];
  const float* bo = (const float*)d_in[8];
  const int*   gi = (const int*)d_in[9];
  float* out = (float*)d_out;

  const size_t NE = (size_t)4 * 16 * 4096 * 64;   // 16.7M floats per buffer
  float* qb  = (float*)d_ws;
  float* kb  = qb + NE;
  float* vb  = kb + NE;
  float* ctx = vb + NE;

  dim3 blk(256);
  dim3 gproj(128, 16);                 // 16384/128 rows x 1024/64 cols
  bigbird_proj_kernel<<<gproj, blk, 0, stream>>>(x, Wq, bq, qb, 1);
  bigbird_proj_kernel<<<gproj, blk, 0, stream>>>(x, Wk, bk, kb, 1);
  bigbird_proj_kernel<<<gproj, blk, 0, stream>>>(x, Wv, bv, vb, 1);

  bigbird_local_kernel<<<dim3(64, 64), blk, 0, stream>>>(qb, kb, vb, ctx);
  bigbird_global_kernel<<<dim3(2, 64), blk, 0, stream>>>(qb, kb, vb, gi, ctx);

  bigbird_proj_kernel<<<gproj, blk, 0, stream>>>(ctx, Wo, bo, out, 0);
}
